// Equi_Nonlin_Grad_Module_71554155151859
// MI455X (gfx1250) — matrix operand order repaired and checked by reading, not compiled, not run on hardware
//
#include <hip/hip_runtime.h>
#include <hip/hip_bf16.h>

// ---------------------------------------------------------------------------
// Equi_Nonlin_Grad: fused bilinear tensor-product head + VJP on gfx1250.
// Heavy contractions -> v_wmma_f32_16x16x32_bf16 GEMMs with double-buffered
// LDS staging via GLOBAL_LOAD_ASYNC_TO_LDS (ASYNCcnt) when available.
// ---------------------------------------------------------------------------

#define NB      2048
#define DIM     480
#define HID     1024
#define ZD      256
#define KCAT    21504        // 128*128 + 64*64 + 32*32
#define OFF1    16384        // start of 64x64 block
#define OFF2    20480        // start of 32x32 block

typedef __attribute__((ext_vector_type(16))) __bf16 v16bf;
typedef __attribute__((ext_vector_type(8)))  float  v8f;

// ---- async global->LDS path (gfx1250), guarded so compile never breaks ----
#if defined(__has_builtin)
#if __has_builtin(__builtin_amdgcn_global_load_async_to_lds_b128) && \
    __has_builtin(__builtin_amdgcn_s_wait_asynccnt)
#define USE_ASYNC_LDS 1
#endif
#endif
#ifndef USE_ASYNC_LDS
#define USE_ASYNC_LDS 0
#endif

#if USE_ASYNC_LDS
typedef int v4i_raw __attribute__((vector_size(16)));
typedef __attribute__((address_space(1))) v4i_raw as1_v4i;
typedef __attribute__((address_space(3))) v4i_raw as3_v4i;
typedef __attribute__((address_space(1))) void    as1_void;
typedef __attribute__((address_space(3))) void    as3_void;
#endif

__device__ __forceinline__ void copy16_g2l(const unsigned short* g,
                                           unsigned short* l) {
#if USE_ASYNC_LDS
    as1_void* g1 = (as1_void*)const_cast<unsigned short*>(g);
    as3_void* l3 = (as3_void*)l;
    __builtin_amdgcn_global_load_async_to_lds_b128(
        (as1_v4i*)g1, (as3_v4i*)l3, 0, 0);
#else
    *reinterpret_cast<uint4*>(l) = *reinterpret_cast<const uint4*>(g);
    __builtin_prefetch(g + 64, 0, 1);   // global_prefetch_b8
#endif
}

__device__ __forceinline__ void wait_async_tiles() {
#if USE_ASYNC_LDS
    __builtin_amdgcn_s_wait_asynccnt(0);
#endif
}

__device__ __forceinline__ unsigned short f2bf(float f) {
    union { float f; unsigned u; } c; c.f = f;
    unsigned u = c.u;
    unsigned r = u + 0x7FFFu + ((u >> 16) & 1u);   // round-to-nearest-even
    return (unsigned short)(r >> 16);
}

// ---------------------------------------------------------------------------
// Packing kernels
// ---------------------------------------------------------------------------
__global__ void f32_to_bf16_kernel(const float* __restrict__ x,
                                   unsigned short* __restrict__ y, size_t n) {
    size_t i = (size_t)blockIdx.x * blockDim.x + threadIdx.x;
    if (i < n) y[i] = f2bf(x[i]);
}

// Wcat[r, h], r in [0,21504): block scales folded in.
__global__ void pack_wcat_kernel(const float* __restrict__ w0,
                                 const float* __restrict__ w1,
                                 const float* __restrict__ w2,
                                 unsigned short* __restrict__ Wc) {
    const float S0 = 0.00681930237f;              // 1/sqrt(21504)
    const float S1 = 0.00393712870f;              // S0/sqrt(3)
    const float S2 = 0.00304968340f;              // S0/sqrt(5)
    size_t g = (size_t)blockIdx.x * blockDim.x + threadIdx.x;  // < 21504*1024
    int h = (int)(g & 1023);
    int r = (int)(g >> 10);
    float v;
    if (r < OFF1)       v = w0[(size_t)r * HID + h] * S0;
    else if (r < OFF2)  v = w1[(size_t)(r - OFF1) * HID + h] * S1;
    else                v = w2[(size_t)(r - OFF2) * HID + h] * S2;
    Wc[g] = f2bf(v);
}

// P[b, idx] = pair products of the three irrep blocks of tensor_in.
__global__ void build_P_kernel(const float* __restrict__ t,
                               unsigned short* __restrict__ P) {
    int b   = blockIdx.y;
    int idx = blockIdx.x * blockDim.x + threadIdx.x;   // < 21504
    const float* tr = t + (size_t)b * DIM;
    float val;
    if (idx < OFF1) {
        int u = idx >> 7, v = idx & 127;
        val = tr[u] * tr[v];
    } else if (idx < OFF2) {
        int loc = idx - OFF1; int u = loc >> 6, v = loc & 63;
        float s = 0.f;
        #pragma unroll
        for (int m = 0; m < 3; ++m) s += tr[128 + u * 3 + m] * tr[128 + v * 3 + m];
        val = s;
    } else {
        int loc = idx - OFF2; int u = loc >> 5, v = loc & 31;
        float s = 0.f;
        #pragma unroll
        for (int m = 0; m < 5; ++m) s += tr[320 + u * 5 + m] * tr[320 + v * 5 + m];
        val = s;
    }
    P[(size_t)b * KCAT + idx] = f2bf(val);
}

// ---------------------------------------------------------------------------
// Tiled bf16 WMMA GEMM:  C(MxN,f32) = A(MxK,bf16) * B(bf16) [+ bias]
// TRANS_B=false: B is (K x N) row-major.  TRANS_B=true: B is (N x K) row-major.
// Block tile 128x128, BK=32, 8 waves of 32x64, double-buffered LDS staging.
// ---------------------------------------------------------------------------
template <bool TRANS_B>
__global__ __launch_bounds__(256) void gemm_bf16_kernel(
        const unsigned short* __restrict__ A,
        const unsigned short* __restrict__ B,
        float* __restrict__ C,
        const float* __restrict__ bias,
        int M, int Nn, int K) {
    constexpr int BM = 128, BN = 128, BK = 32, S = BK + 8;   // S=40 shorts
    __shared__ alignas(16) unsigned short As[2][BM * S];
    __shared__ alignas(16) unsigned short Bs[2][BN * S];     // stored [n][k]

    const int tid  = threadIdx.x;
    const int bm   = blockIdx.y * BM;
    const int bn   = blockIdx.x * BN;
    const int lane = tid & 31;
    const int wave = tid >> 5;
    const int wm   = wave & 3;        // 4 row groups of 32
    const int wn   = wave >> 2;       // 2 col groups of 64
    const int r    = lane & 15;
    const int half = lane >> 4;

    v8f acc[2][4];
    #pragma unroll
    for (int i = 0; i < 2; ++i)
        #pragma unroll
        for (int j = 0; j < 4; ++j)
            acc[i][j] = (v8f){0.f, 0.f, 0.f, 0.f, 0.f, 0.f, 0.f, 0.f};

    const int aRow  = tid >> 1;          // 0..127
    const int aCol  = (tid & 1) * 16;    // 0 / 16
    const int bRowK = tid >> 3;          // non-trans: k 0..31
    const int bSeg  = (tid & 7) * 16;    // non-trans: n offset
    const int btN   = tid >> 1;          // trans: n 0..127
    const int btK   = (tid & 1) * 16;    // trans: k offset

    // stage one (A,B) tile pair for k-offset k0 into buffer buf
    auto stage = [&](int buf, int k0) {
        copy16_g2l(A + (size_t)(bm + aRow) * K + k0 + aCol,
                   &As[buf][aRow * S + aCol]);
        if (TRANS_B) {
            copy16_g2l(B + (size_t)(bn + btN) * K + k0 + btK,
                       &Bs[buf][btN * S + btK]);
        } else {
            const unsigned short* src = B + (size_t)(k0 + bRowK) * Nn + bn + bSeg;
            uint4 d = *reinterpret_cast<const uint4*>(src);
            const unsigned short* d16 = reinterpret_cast<const unsigned short*>(&d);
            #pragma unroll
            for (int e = 0; e < 16; ++e)
                Bs[buf][(bSeg + e) * S + bRowK] = d16[e];
        }
    };

    const int nk = K / BK;
    stage(0, 0);                                   // prologue
    for (int i = 0; i < nk; ++i) {
        const int cur = i & 1;
        wait_async_tiles();                        // s_wait_asynccnt 0
        __syncthreads();                           // tile `cur` visible to all
        if (i + 1 < nk)
            stage(cur ^ 1, (i + 1) * BK);          // overlap next tile load

        // ---- build fragments per ISA VGPR layouts ----
        union Frag { unsigned int u[8]; v16bf v; };
        Frag af[2], bf[4];
        #pragma unroll
        for (int mi = 0; mi < 2; ++mi) {
            int row = wm * 32 + mi * 16 + r;
            #pragma unroll
            for (int j = 0; j < 8; ++j) {
                // A 16x32 bf16: lanes0-15 K {0..7,16..23}; lanes16-31 K {8..15,24..31}
                int k = (j < 4) ? (half * 8 + 2 * j) : (16 + half * 8 + 2 * (j - 4));
                af[mi].u[j] =
                    *reinterpret_cast<const unsigned int*>(&As[cur][row * S + k]);
            }
        }
        #pragma unroll
        for (int ni = 0; ni < 4; ++ni) {
            int col = wn * 64 + ni * 16 + r;
            #pragma unroll
            for (int j = 0; j < 8; ++j) {
                // B 32x16 bf16: lanes0-15 K {0..7,16..23}; lanes16-31 K {8..15,24..31}
                int k = (j < 4) ? (half * 8 + 2 * j) : (16 + half * 8 + 2 * (j - 4));
                bf[ni].u[j] =
                    *reinterpret_cast<const unsigned int*>(&Bs[cur][col * S + k]);
            }
        }
        // ---- 8 WMMA per k-step ----
        #pragma unroll
        for (int mi = 0; mi < 2; ++mi)
            #pragma unroll
            for (int ni = 0; ni < 4; ++ni)
                acc[mi][ni] = __builtin_amdgcn_wmma_f32_16x16x32_bf16(
                    false, af[mi].v, false, bf[ni].v,
                    (short)0, acc[mi][ni], false, false);
        __syncthreads();                           // reads of `cur` done
    }

    // ---- epilogue: C/D layout row = half*8 + vgpr, col = lane%16 ----
    #pragma unroll
    for (int mi = 0; mi < 2; ++mi)
        #pragma unroll
        for (int ni = 0; ni < 4; ++ni) {
            int colg = bn + wn * 64 + ni * 16 + r;
            float bv = bias ? bias[colg] : 0.f;
            #pragma unroll
            for (int j = 0; j < 8; ++j) {
                int rowg = bm + wm * 32 + mi * 16 + half * 8 + j;
                C[(size_t)rowg * Nn + colg] = acc[mi][ni][j] + bv;
            }
        }
}

// ---------------------------------------------------------------------------
// LayerNorm + SiLU forward (stores pre-silu L, post-silu A, row stats)
// ---------------------------------------------------------------------------
__device__ __forceinline__ float blockReduceSum(float v, float* sbuf) {
    int tid = threadIdx.x;
    sbuf[tid] = v; __syncthreads();
    for (int s = 128; s > 0; s >>= 1) {
        if (tid < s) sbuf[tid] += sbuf[tid + s];
        __syncthreads();
    }
    float r = sbuf[0]; __syncthreads();
    return r;
}

__global__ __launch_bounds__(256) void ln_silu_fwd_kernel(
        const float* __restrict__ Z, const float* __restrict__ g,
        const float* __restrict__ be, float* __restrict__ L,
        float* __restrict__ A, float* __restrict__ mu_o,
        float* __restrict__ rs_o, int D) {
    __shared__ float red[256];
    int row = blockIdx.x;
    const float* z = Z + (size_t)row * D;
    float s1 = 0.f, s2 = 0.f;
    for (int i = threadIdx.x; i < D; i += 256) { float v = z[i]; s1 += v; s2 += v * v; }
    s1 = blockReduceSum(s1, red);
    s2 = blockReduceSum(s2, red);
    float mu  = s1 / D;
    float var = s2 / D - mu * mu;
    float rs  = rsqrtf(var + 1e-6f);
    if (threadIdx.x == 0) { mu_o[row] = mu; rs_o[row] = rs; }
    for (int i = threadIdx.x; i < D; i += 256) {
        float xh = (z[i] - mu) * rs;
        float l  = xh * g[i] + be[i];
        L[(size_t)row * D + i] = l;
        float sg = 1.f / (1.f + __expf(-l));
        A[(size_t)row * D + i] = l * sg;
    }
}

// SiLU' * upstream, then LayerNorm backward -> dZ (pre-LN gradient).
// dA == nullptr means upstream cotangent is all-ones.
__global__ __launch_bounds__(256) void ln_silu_bwd_kernel(
        const float* __restrict__ dA, const float* __restrict__ L,
        const float* __restrict__ Z, const float* __restrict__ mu_i,
        const float* __restrict__ rs_i, const float* __restrict__ g,
        float* __restrict__ dZ, int D) {
    __shared__ float red[256];
    int row = blockIdx.x;
    float mu = mu_i[row], rs = rs_i[row];
    const float* l = L + (size_t)row * D;
    const float* z = Z + (size_t)row * D;
    float t1 = 0.f, t2 = 0.f;
    for (int i = threadIdx.x; i < D; i += 256) {
        float lv = l[i];
        float sg = 1.f / (1.f + __expf(-lv));
        float up = dA ? dA[(size_t)row * D + i] : 1.f;
        float dl = up * sg * (1.f + lv * (1.f - sg));
        float xh = (z[i] - mu) * rs;
        float dxh = dl * g[i];
        t1 += dxh; t2 += dxh * xh;
    }
    t1 = blockReduceSum(t1, red) / D;
    t2 = blockReduceSum(t2, red) / D;
    for (int i = threadIdx.x; i < D; i += 256) {
        float lv = l[i];
        float sg = 1.f / (1.f + __expf(-lv));
        float up = dA ? dA[(size_t)row * D + i] : 1.f;
        float dl = up * sg * (1.f + lv * (1.f - sg));
        float xh = (z[i] - mu) * rs;
        float dxh = dl * g[i];
        dZ[(size_t)row * D + i] = rs * (dxh - t1 - xh * t2);
    }
}

// ---------------------------------------------------------------------------
// fctp input-gradient:  dx[b,u] = sum_v x[b,v] * (Q[b,u,v] + Q[b,v,u])
// Q[b,:] = dh0[b] contracted with Wcat (scales already folded into Wcat).
// ---------------------------------------------------------------------------
__global__ __launch_bounds__(128) void fctp_grad_kernel(
        const float* __restrict__ t, const float* __restrict__ Q,
        float* __restrict__ out) {
    int b = blockIdx.x;
    const float* tr = t + (size_t)b * DIM;
    const float* q  = Q + (size_t)b * KCAT;
    float* o = out + (size_t)b * DIM;
    for (int oidx = threadIdx.x; oidx < DIM; oidx += blockDim.x) {
        float acc = 0.f;
        if (oidx < 128) {
            int u = oidx;
            for (int v = 0; v < 128; ++v)
                acc += tr[v] * (q[u * 128 + v] + q[v * 128 + u]);
        } else if (oidx < 320) {
            int loc = oidx - 128; int u = loc / 3, m = loc % 3;
            const float* q1 = q + OFF1;
            for (int v = 0; v < 64; ++v)
                acc += tr[128 + v * 3 + m] * (q1[u * 64 + v] + q1[v * 64 + u]);
        } else {
            int loc = oidx - 320; int u = loc / 5, m = loc % 5;
            const float* q2 = q + OFF2;
            for (int v = 0; v < 32; ++v)
                acc += tr[320 + v * 5 + m] * (q2[u * 32 + v] + q2[v * 32 + u]);
        }
        o[oidx] = acc;
    }
}

// ---------------------------------------------------------------------------
// Host orchestration
// ---------------------------------------------------------------------------
extern "C" void kernel_launch(void* const* d_in, const int* in_sizes, int n_in,
                              void* d_out, int out_size, void* d_ws, size_t ws_size,
                              hipStream_t stream) {
    const float* t   = (const float*)d_in[0];
    const float* w0  = (const float*)d_in[1];
    const float* w1  = (const float*)d_in[2];
    const float* w2  = (const float*)d_in[3];
    const float* W1  = (const float*)d_in[4];
    const float* b1  = (const float*)d_in[5];
    const float* g1  = (const float*)d_in[6];
    const float* be1 = (const float*)d_in[7];
    const float* W2  = (const float*)d_in[8];
    const float* b2  = (const float*)d_in[9];
    const float* g2  = (const float*)d_in[10];
    const float* be2 = (const float*)d_in[11];
    float* out_fwd  = (float*)d_out;                      // (N, Z)
    float* out_grad = (float*)d_out + (size_t)NB * ZD;    // (N, DIM)

    // workspace bump allocator (256B aligned)
    size_t off = 0;
    auto alloc = [&](size_t bytes) -> void* {
        void* p = (char*)d_ws + off;
        off += (bytes + 255) & ~(size_t)255;
        return p;
    };
    unsigned short* Pbf    = (unsigned short*)alloc((size_t)NB * KCAT * 2);
    unsigned short* Wcat   = (unsigned short*)alloc((size_t)KCAT * HID * 2);
    unsigned short* W1bf   = (unsigned short*)alloc((size_t)HID * HID * 2);
    unsigned short* W2bf   = (unsigned short*)alloc((size_t)HID * ZD * 2);
    float* H0    = (float*)alloc((size_t)NB * HID * 4);
    unsigned short* H0bf   = (unsigned short*)alloc((size_t)NB * HID * 2);
    float* Z1    = (float*)alloc((size_t)NB * HID * 4);
    float* L1    = (float*)alloc((size_t)NB * HID * 4);
    float* A1    = (float*)alloc((size_t)NB * HID * 4);
    unsigned short* A1bf   = (unsigned short*)alloc((size_t)NB * HID * 2);
    float* mu1   = (float*)alloc((size_t)NB * 4);
    float* rs1   = (float*)alloc((size_t)NB * 4);
    float* Z2    = (float*)alloc((size_t)NB * ZD * 4);
    float* L2    = (float*)alloc((size_t)NB * ZD * 4);
    float* mu2   = (float*)alloc((size_t)NB * 4);
    float* rs2   = (float*)alloc((size_t)NB * 4);
    float* dZ2   = (float*)alloc((size_t)NB * ZD * 4);
    unsigned short* dZ2bf  = (unsigned short*)alloc((size_t)NB * ZD * 2);
    float* dA1   = (float*)alloc((size_t)NB * HID * 4);
    float* dZ1   = (float*)alloc((size_t)NB * HID * 4);
    unsigned short* dZ1bf  = (unsigned short*)alloc((size_t)NB * HID * 2);
    float* dH0   = (float*)alloc((size_t)NB * HID * 4);
    unsigned short* dH0bf  = (unsigned short*)alloc((size_t)NB * HID * 2);
    float* Q     = (float*)alloc((size_t)NB * KCAT * 4);
    (void)ws_size; (void)n_in; (void)in_sizes; (void)out_size;

    const dim3 blk256(256);

    // ---- pack weights / activations to bf16 ----
    pack_wcat_kernel<<<dim3((KCAT * HID) / 256), blk256, 0, stream>>>(w0, w1, w2, Wcat);
    build_P_kernel<<<dim3(KCAT / 256, NB), blk256, 0, stream>>>(t, Pbf);
    f32_to_bf16_kernel<<<dim3((HID * HID) / 256), blk256, 0, stream>>>(W1, W1bf, (size_t)HID * HID);
    f32_to_bf16_kernel<<<dim3((HID * ZD) / 256), blk256, 0, stream>>>(W2, W2bf, (size_t)HID * ZD);

    // ---- forward ----
    // H0 = P @ Wcat                        (2048 x 21504) x (21504 x 1024)
    gemm_bf16_kernel<false><<<dim3(HID / 128, NB / 128), blk256, 0, stream>>>(
        Pbf, Wcat, H0, nullptr, NB, HID, KCAT);
    f32_to_bf16_kernel<<<dim3((NB * HID) / 256), blk256, 0, stream>>>(H0, H0bf, (size_t)NB * HID);
    // Z1 = H0 @ W1 + b1
    gemm_bf16_kernel<false><<<dim3(HID / 128, NB / 128), blk256, 0, stream>>>(
        H0bf, W1bf, Z1, b1, NB, HID, HID);
    ln_silu_fwd_kernel<<<dim3(NB), blk256, 0, stream>>>(Z1, g1, be1, L1, A1, mu1, rs1, HID);
    f32_to_bf16_kernel<<<dim3((NB * HID) / 256), blk256, 0, stream>>>(A1, A1bf, (size_t)NB * HID);
    // Z2 = A1 @ W2 + b2
    gemm_bf16_kernel<false><<<dim3(ZD / 128, NB / 128), blk256, 0, stream>>>(
        A1bf, W2bf, Z2, b2, NB, ZD, HID);
    // out = silu(ln(Z2))  (A-output written straight into d_out)
    ln_silu_fwd_kernel<<<dim3(NB), blk256, 0, stream>>>(Z2, g2, be2, L2, out_fwd, mu2, rs2, ZD);

    // ---- backward (cotangent = ones) ----
    ln_silu_bwd_kernel<<<dim3(NB), blk256, 0, stream>>>(nullptr, L2, Z2, mu2, rs2, g2, dZ2, ZD);
    f32_to_bf16_kernel<<<dim3((NB * ZD) / 256), blk256, 0, stream>>>(dZ2, dZ2bf, (size_t)NB * ZD);
    // dA1 = dZ2 @ W2^T                     (TRANS_B, B = W2 (1024 x 256))
    gemm_bf16_kernel<true><<<dim3(HID / 128, NB / 128), blk256, 0, stream>>>(
        dZ2bf, W2bf, dA1, nullptr, NB, HID, ZD);
    ln_silu_bwd_kernel<<<dim3(NB), blk256, 0, stream>>>(dA1, L1, Z1, mu1, rs1, g1, dZ1, HID);
    f32_to_bf16_kernel<<<dim3((NB * HID) / 256), blk256, 0, stream>>>(dZ1, dZ1bf, (size_t)NB * HID);
    // dH0 = dZ1 @ W1^T
    gemm_bf16_kernel<true><<<dim3(HID / 128, NB / 128), blk256, 0, stream>>>(
        dZ1bf, W1bf, dH0, nullptr, NB, HID, HID);
    f32_to_bf16_kernel<<<dim3((NB * HID) / 256), blk256, 0, stream>>>(dH0, dH0bf, (size_t)NB * HID);
    // Q = dH0 @ Wcat^T                     (2048 x 1024) x (1024 x 21504)
    gemm_bf16_kernel<true><<<dim3(KCAT / 128, NB / 128), blk256, 0, stream>>>(
        dH0bf, Wcat, Q, nullptr, NB, KCAT, HID);
    // dx = per-sample symmetric contraction
    fctp_grad_kernel<<<dim3(NB), dim3(128), 0, stream>>>(t, Q, out_grad);
}
